// GraphUnet_8143257993639
// MI455X (gfx1250) — 8-bit matrix operand order repaired and checked by reading, compiled, not run on hardware
//
#include <hip/hip_runtime.h>
#include <hip/hip_bf16.h>

typedef float v2f __attribute__((ext_vector_type(2)));
typedef float v8f __attribute__((ext_vector_type(8)));
typedef int   v8i __attribute__((ext_vector_type(8)));

#define DIM 256

__device__ __forceinline__ unsigned lds_off(const void* p) {
  return (unsigned)(unsigned long long)(__attribute__((address_space(3))) const char*)p;
}

template <bool TRANSB>
__global__ __launch_bounds__(128) void gemm_f32_wmma(
    const float* __restrict__ A, int lda, const float* __restrict__ B, int ldb,
    float* __restrict__ C, int ldc, const float* __restrict__ bias,
    int M, int N, int K) {
  __shared__ float As[2][64][20];
  __shared__ float Bs[2][16][68];
  const int tid  = threadIdx.x;
  const int wave = tid >> 5;
  const int lane = tid & 31;
  const int wm   = (wave >> 1) * 32;
  const int wn   = (wave & 1) * 32;
  const int bm   = blockIdx.y * 64;
  const int bn   = blockIdx.x * 64;
  const int lhalf = lane >> 4;
  const int l16   = lane & 15;

  const bool interior = (bm + 64 <= M) && (bn + 64 <= N);
  const unsigned aBase = lds_off(&As[0][0][0]);
  const unsigned bBase = lds_off(&Bs[0][0][0]);
  const int S = (K + 15) >> 4;

  v8f acc[2][2] = {};

  auto stage = [&](int buf, int s) {
    const int k0 = s << 4;
    if (interior && (k0 + 16) <= K) {
#pragma unroll
      for (int it = 0; it < 8; ++it) {
        const int t = tid + it * 128;
        {
          const int r = t >> 4, c = t & 15;
          unsigned lds = aBase + (unsigned)(buf * (64 * 20) + r * 20 + c) * 4u;
          unsigned off = (unsigned)((bm + r) * lda + (k0 + c)) * 4u;
          asm volatile("global_load_async_to_lds_b32 %0, %1, %2 offset:0"
                       :: "v"(lds), "v"(off), "s"(A)
                       : "memory");
        }
        {
          const int r = t >> 6, c = t & 63;
          unsigned lds = bBase + (unsigned)(buf * (16 * 68) + r * 68 + c) * 4u;
          unsigned off = TRANSB ? (unsigned)((bn + c) * ldb + (k0 + r)) * 4u
                                : (unsigned)((k0 + r) * ldb + (bn + c)) * 4u;
          asm volatile("global_load_async_to_lds_b32 %0, %1, %2 offset:0"
                       :: "v"(lds), "v"(off), "s"(B)
                       : "memory");
        }
      }
    } else {
      for (int t = tid; t < 64 * 16; t += 128) {
        int r = t >> 4, c = t & 15;
        int gr = bm + r, gc = k0 + c;
        As[buf][r][c] = (gr < M && gc < K) ? A[(size_t)gr * lda + gc] : 0.f;
      }
      for (int t = tid; t < 16 * 64; t += 128) {
        int r = t >> 6, c = t & 63;
        int gk = k0 + r, gn = bn + c;
        float vb = 0.f;
        if (gk < K && gn < N)
          vb = TRANSB ? B[(size_t)gn * ldb + gk] : B[(size_t)gk * ldb + gn];
        Bs[buf][r][c] = vb;
      }
    }
  };

  stage(0, 0);
  int buf = 0;
  for (int s = 0; s < S; ++s) {
    const bool nextAsync =
        (s + 1 < S) && interior && (((s + 1) << 4) + 16 <= K);
    if (s + 1 < S) stage(buf ^ 1, s + 1);
    if (nextAsync)
      asm volatile("s_wait_asynccnt 16" ::: "memory");
    else
      asm volatile("s_wait_asynccnt 0" ::: "memory");
    __syncthreads();
#pragma unroll
    for (int kk = 0; kk < 16; kk += 4) {
      v2f af[2], bf[2];
#pragma unroll
      for (int i = 0; i < 2; ++i) {
        int row = wm + i * 16 + l16;
        af[i].x = As[buf][row][kk + 2 * lhalf];
        af[i].y = As[buf][row][kk + 2 * lhalf + 1];
        int col = wn + i * 16 + l16;
        bf[i].x = Bs[buf][kk + 2 * lhalf][col];
        bf[i].y = Bs[buf][kk + 2 * lhalf + 1][col];
      }
#pragma unroll
      for (int i = 0; i < 2; ++i)
#pragma unroll
        for (int j = 0; j < 2; ++j)
          acc[i][j] = __builtin_amdgcn_wmma_f32_16x16x4_f32(
              false, af[i], false, bf[j], (short)0, acc[i][j], false, false);
    }
    __syncthreads();
    buf ^= 1;
  }
#pragma unroll
  for (int i = 0; i < 2; ++i)
#pragma unroll
    for (int j = 0; j < 2; ++j)
#pragma unroll
      for (int r = 0; r < 8; ++r) {
        int row = bm + wm + i * 16 + r + 8 * lhalf;
        int col = bn + wn + j * 16 + l16;
        if (row < M && col < N) {
          float vv = acc[i][j][r];
          if (bias) vv += bias[col];
          C[(size_t)row * ldc + col] = vv;
        }
      }
}

__global__ __launch_bounds__(128) void bgemm_iu8_wmma(
    const float* __restrict__ g, int ldg, const int* __restrict__ idx,
    float* __restrict__ out, int ldo, int M, int N, int K) {
  __shared__ int AsI[64][17];
  __shared__ int BsI[64][17];
  const int tid  = threadIdx.x;
  const int wave = tid >> 5;
  const int lane = tid & 31;
  const int wm   = (wave >> 1) * 32;
  const int wn   = (wave & 1) * 32;
  const int bm   = blockIdx.y * 64;
  const int bn   = blockIdx.x * 64;
  const int lhalf = lane >> 4;
  const int l16   = lane & 15;

  v8i acc[2][2] = {};

  for (int k0 = 0; k0 < K; k0 += 64) {
    for (int t = tid; t < 64 * 16; t += 128) {
      int r = t >> 4, q = t & 15;
      int gr = bm + r;
      unsigned pk = 0;
      if (gr < M) {
        int grow = idx ? idx[gr] : gr;
#pragma unroll
        for (int bb = 0; bb < 4; ++bb) {
          int gk = k0 + 4 * q + bb;
          if (gk < K && g[(size_t)grow * ldg + gk] != 0.f) pk |= (1u << (8 * bb));
        }
      }
      AsI[r][q] = (int)pk;
    }
    for (int t = tid; t < 64 * 16; t += 128) {
      int c = t >> 4, q = t & 15;
      int gn = bn + c;
      unsigned pk = 0;
      if (gn < N) {
#pragma unroll
        for (int bb = 0; bb < 4; ++bb) {
          int gk = k0 + 4 * q + bb;
          if (gk < K && g[(size_t)gk * ldg + gn] != 0.f) pk |= (1u << (8 * bb));
        }
      }
      BsI[c][q] = (int)pk;
    }
    __syncthreads();
    v8i af[2], bf[2];
#pragma unroll
    for (int i = 0; i < 2; ++i) {
      int row  = wm + i * 16 + l16;
      int coln = wn + i * 16 + l16;
#pragma unroll
      for (int v = 0; v < 8; ++v) {
        int kqA = ((v >> 1) * 16 + (v & 1) * 4 + lhalf * 8) >> 2;
        af[i][v] = AsI[row][kqA];
        int kqB = ((v >> 1) * 16 + (v & 1) * 4 + lhalf * 8) >> 2;
        bf[i][v] = BsI[coln][kqB];
      }
    }
#pragma unroll
    for (int i = 0; i < 2; ++i)
#pragma unroll
      for (int j = 0; j < 2; ++j)
        acc[i][j] = __builtin_amdgcn_wmma_i32_16x16x64_iu8(
            false, af[i], false, bf[j], acc[i][j], false, false);
    __syncthreads();
  }
#pragma unroll
  for (int i = 0; i < 2; ++i)
#pragma unroll
    for (int j = 0; j < 2; ++j)
#pragma unroll
      for (int r = 0; r < 8; ++r) {
        int row = bm + wm + i * 16 + r + 8 * lhalf;
        int col = bn + wn + j * 16 + l16;
        if (row < M && col < N)
          out[(size_t)row * ldo + col] = (acc[i][j][r] != 0) ? 1.0f : 0.0f;
      }
}

__global__ void bn_kernel(const float* __restrict__ x,
                          const float* __restrict__ gamma,
                          const float* __restrict__ beta,
                          float* __restrict__ y, int n) {
  int col = blockIdx.x;
  __shared__ float red[256];
  float s = 0.f;
  for (int i = threadIdx.x; i < n; i += 256) s += x[(size_t)i * DIM + col];
  red[threadIdx.x] = s;
  __syncthreads();
  for (int o = 128; o; o >>= 1) {
    if (threadIdx.x < o) red[threadIdx.x] += red[threadIdx.x + o];
    __syncthreads();
  }
  float mean = red[0] / n;
  __syncthreads();
  float sv = 0.f;
  for (int i = threadIdx.x; i < n; i += 256) {
    float d = x[(size_t)i * DIM + col] - mean;
    sv += d * d;
  }
  red[threadIdx.x] = sv;
  __syncthreads();
  for (int o = 128; o; o >>= 1) {
    if (threadIdx.x < o) red[threadIdx.x] += red[threadIdx.x + o];
    __syncthreads();
  }
  float inv = rsqrtf(red[0] / n + 1e-5f);
  float ga = gamma[col], be = beta[col];
  for (int i = threadIdx.x; i < n; i += 256)
    y[(size_t)i * DIM + col] = ga * (x[(size_t)i * DIM + col] - mean) * inv + be;
}

__global__ void rowsum_kernel(const float* __restrict__ m, int ld, int w,
                              float* __restrict__ out) {
  int row = blockIdx.x;
  __shared__ float red[256];
  float s = 0.f;
  for (int j = threadIdx.x; j < w; j += 256) s += m[(size_t)row * ld + j];
  red[threadIdx.x] = s;
  __syncthreads();
  for (int o = 128; o; o >>= 1) {
    if (threadIdx.x < o) red[threadIdx.x] += red[threadIdx.x + o];
    __syncthreads();
  }
  if (threadIdx.x == 0) out[row] = red[0];
}

__global__ void vec_rsqrt_p1_kernel(const float* in, float* out, int n) {
  int i = blockIdx.x * 256 + threadIdx.x;
  if (i < n) out[i] = rsqrtf(in[i] + 1.0f);
}
__global__ void vec_recip_kernel(const float* in, float* out, int n) {
  int i = blockIdx.x * 256 + threadIdx.x;
  if (i < n) out[i] = 1.0f / in[i];
}

__global__ void rowscale_kernel(const float* __restrict__ in,
                                const float* __restrict__ s,
                                float* __restrict__ out, int n) {
  int i = blockIdx.x, j = threadIdx.x;
  out[(size_t)i * DIM + j] = in[(size_t)i * DIM + j] * s[i];
}

__global__ void gcn_epi_kernel(const float* __restrict__ v,
                               const float* __restrict__ u,
                               const float* __restrict__ d,
                               float* __restrict__ out, int n) {
  int i = blockIdx.x, j = threadIdx.x;
  float x = d[i] * (v[(size_t)i * DIM + j] + u[(size_t)i * DIM + j]);
  out[(size_t)i * DIM + j] = (x >= 0.f) ? x : 0.01f * x;
}

__global__ void pool_row_kernel(const float* __restrict__ hb,
                                const float* __restrict__ gh,
                                const float* __restrict__ rs,
                                const float* __restrict__ pw,
                                const float* __restrict__ pb_arr, int level,
                                float* __restrict__ pl_raw,
                                float* __restrict__ lp,
                                float* __restrict__ pf_raw) {
  int i = blockIdx.x, j = threadIdx.x;
  __shared__ float r1[256], r2[256], r3[256];
  float hv = hb[(size_t)i * DIM + j];
  float gv = gh[(size_t)i * DIM + j];
  float w = pw[j];
  r1[j] = fabsf(hv - gv);
  r2[j] = gv * w;
  r3[j] = hv * w;
  __syncthreads();
  for (int o = 128; o; o >>= 1) {
    if (j < o) { r1[j] += r1[j + o]; r2[j] += r2[j + o]; r3[j] += r3[j + o]; }
    __syncthreads();
  }
  if (j == 0) {
    float pb = pb_arr[level];
    pl_raw[i] = r1[0] + rs[i];
    lp[i] = r2[0] + pb;
    pf_raw[i] = r3[0] + pb;
  }
}

__global__ void pool_scores_kernel(const float* __restrict__ pl_raw,
                                   const float* __restrict__ lp,
                                   const float* __restrict__ pf_raw,
                                   const float* __restrict__ sigma_arr,
                                   int level, int n, float* __restrict__ scores) {
  __shared__ float red[256];
  __shared__ float s_mx, s_sum;
  float m = -3.4e38f;
  for (int i = threadIdx.x; i < n; i += 256) m = fmaxf(m, lp[i]);
  red[threadIdx.x] = m;
  __syncthreads();
  for (int o = 128; o; o >>= 1) {
    if (threadIdx.x < o) red[threadIdx.x] = fmaxf(red[threadIdx.x], red[threadIdx.x + o]);
    __syncthreads();
  }
  if (threadIdx.x == 0) s_mx = red[0];
  __syncthreads();
  float mx = s_mx;
  float s = 0.f;
  for (int i = threadIdx.x; i < n; i += 256) s += __expf(lp[i] - mx);
  red[threadIdx.x] = s;
  __syncthreads();
  for (int o = 128; o; o >>= 1) {
    if (threadIdx.x < o) red[threadIdx.x] += red[threadIdx.x + o];
    __syncthreads();
  }
  if (threadIdx.x == 0) s_sum = red[0];
  __syncthreads();
  float inv = 1.f / s_sum;
  float sg = sigma_arr[level];
  for (int i = threadIdx.x; i < n; i += 256) {
    float pg = __expf(lp[i] - mx) * inv;
    float pl = 1.f / (1.f + __expf(-pl_raw[i]));
    float pt = 1.f / (1.f + __expf(-(pl + pg)));
    float pf = 1.f / (1.f + __expf(-pf_raw[i]));
    scores[i] = sg * pt + (1.f - sg) * pf;
  }
}

__global__ void topk_kernel(const float* __restrict__ s, int n, int kk,
                            int* __restrict__ idx) {
  int i = blockIdx.x * 256 + threadIdx.x;
  if (i >= n) return;
  float si = s[i];
  int rank = 0;
  for (int j = 0; j < n; ++j) {
    float sj = s[j];
    rank += (sj > si) || (sj == si && j < i);
  }
  if (rank < kk) idx[rank] = i;
}

__global__ void gather_cols_kernel(const float* __restrict__ ahat, int lda,
                                   const int* __restrict__ idx,
                                   float* __restrict__ out, int ldo, int cols) {
  int c = blockIdx.x * 256 + threadIdx.x;
  int r = blockIdx.y;
  if (c < cols) out[(size_t)r * ldo + c] = ahat[(size_t)r * lda + idx[c]];
}

__global__ void coldiv_kernel(const float* __restrict__ in, int ldi,
                              float* __restrict__ out, int ldo, int cols,
                              const float* __restrict__ s) {
  int c = blockIdx.x * 256 + threadIdx.x;
  int r = blockIdx.y;
  if (c < cols) out[(size_t)r * ldo + c] = in[(size_t)r * ldi + c] / s[c];
}

__global__ void fill_zero_kernel(float* p, int count) {
  int i = blockIdx.x * 256 + threadIdx.x;
  if (i < count) p[i] = 0.f;
}

__global__ void scatter_rows_kernel(const float* __restrict__ h_in,
                                    const int* __restrict__ idx,
                                    float* __restrict__ h_out) {
  int r = blockIdx.x, j = threadIdx.x;
  h_out[(size_t)idx[r] * DIM + j] = h_in[(size_t)r * DIM + j];
}

__global__ void softmax_row_kernel(float* __restrict__ x, int ld, int w) {
  int row = blockIdx.x;
  float* p = x + (size_t)row * ld;
  __shared__ float red[256];
  __shared__ float sh;
  float m = -3.4e38f;
  for (int j = threadIdx.x; j < w; j += 256) m = fmaxf(m, p[j]);
  red[threadIdx.x] = m;
  __syncthreads();
  for (int o = 128; o; o >>= 1) {
    if (threadIdx.x < o) red[threadIdx.x] = fmaxf(red[threadIdx.x], red[threadIdx.x + o]);
    __syncthreads();
  }
  if (threadIdx.x == 0) sh = red[0];
  __syncthreads();
  float mx = sh;
  float s = 0.f;
  for (int j = threadIdx.x; j < w; j += 256) s += __expf(p[j] - mx);
  red[threadIdx.x] = s;
  __syncthreads();
  for (int o = 128; o; o >>= 1) {
    if (threadIdx.x < o) red[threadIdx.x] += red[threadIdx.x + o];
    __syncthreads();
  }
  if (threadIdx.x == 0) sh = red[0];
  __syncthreads();
  float inv = 1.f / sh;
  for (int j = threadIdx.x; j < w; j += 256) p[j] = __expf(p[j] - mx) * inv;
}

__global__ void cnn_combine_kernel(const float* __restrict__ sm,
                                   const float* __restrict__ h3,
                                   const float* __restrict__ h2,
                                   float* __restrict__ out, int n) {
  int i = blockIdx.x, j = threadIdx.x;
  size_t o = (size_t)i * DIM + j;
  out[o] = sm[o] * h3[o] + h2[o] + h3[o];
}

__global__ void add_kernel(const float* __restrict__ a,
                           const float* __restrict__ b,
                           float* __restrict__ out) {
  int i = blockIdx.x, j = threadIdx.x;
  size_t o = (size_t)i * DIM + j;
  out[o] = a[o] + b[o];
}

extern "C" void kernel_launch(void* const* d_in, const int* in_sizes, int n_in,
                              void* d_out, int out_size, void* d_ws,
                              size_t ws_size, hipStream_t stream) {
  (void)in_sizes; (void)n_in; (void)out_size; (void)ws_size;
  const int N0 = 3072, KK0 = 2457, KK1 = 1474;

  const float* g0 = (const float*)d_in[0];
  const float* h0 = (const float*)d_in[1];
  const float* down_bn_g = (const float*)d_in[2];
  const float* down_bn_b = (const float*)d_in[3];
  const float* down_W = (const float*)d_in[4];
  const float* down_b = (const float*)d_in[5];
  const float* bot_bn_g = (const float*)d_in[6];
  const float* bot_bn_b = (const float*)d_in[7];
  const float* bot_W = (const float*)d_in[8];
  const float* bot_b = (const float*)d_in[9];
  const float* up_bn_g = (const float*)d_in[10];
  const float* up_bn_b = (const float*)d_in[11];
  const float* up_W = (const float*)d_in[12];
  const float* up_b = (const float*)d_in[13];
  const float* pool_bn_g = (const float*)d_in[14];
  const float* pool_bn_b = (const float*)d_in[15];
  const float* pool_W = (const float*)d_in[16];
  const float* pool_b = (const float*)d_in[17];
  const float* pool_sigma = (const float*)d_in[18];
  const float* cnn_bn_g = (const float*)d_in[19];
  const float* cnn_bn_b = (const float*)d_in[20];
  const float* cnn_W = (const float*)d_in[21];
  const float* cnn_b = (const float*)d_in[22];

  char* ws = (char*)d_ws;
  size_t off = 0;
  auto alloc = [&](size_t bytes) -> void* {
    off = (off + 255) & ~(size_t)255;
    void* p = ws + off;
    off += bytes;
    return p;
  };
  const size_t HB = (size_t)N0 * DIM * sizeof(float);
  float* h_down0 = (float*)alloc(HB);
  float* h_down1 = (float*)alloc(HB);
  float* g1 = (float*)alloc((size_t)KK0 * KK0 * sizeof(float));
  float* g2 = (float*)alloc((size_t)KK1 * KK1 * sizeof(float));
  int* idx0 = (int*)alloc(KK0 * sizeof(int));
  int* idx1 = (int*)alloc(KK1 * sizeof(int));
  float* h_cur = (float*)alloc(HB);
  float* bufA = (float*)alloc(HB);
  float* bufB = (float*)alloc(HB);
  float* bufC = (float*)alloc(HB);
  float* bufD = (float*)alloc(HB);
  float* bufE = (float*)alloc(HB);
  float* bufF = (float*)alloc(HB);
  float* bufG = (float*)alloc(HB);
  float* vec1 = (float*)alloc(N0 * sizeof(float));
  float* vec2 = (float*)alloc(N0 * sizeof(float));
  float* vec3 = (float*)alloc(N0 * sizeof(float));
  float* vec4 = (float*)alloc(N0 * sizeof(float));
  float* vec5 = (float*)alloc(N0 * sizeof(float));
  float* vec6 = (float*)alloc(N0 * sizeof(float));
  float* big1 = (float*)alloc((size_t)N0 * N0 * sizeof(float));
  float* big2 = (float*)alloc((size_t)KK0 * KK0 * sizeof(float));

  auto gemm = [&](const float* A, int lda, const float* B, int ldb, float* C,
                  int ldc, const float* bias, int M, int Nn, int K, bool tB) {
    dim3 grid((Nn + 63) / 64, (M + 63) / 64);
    if (tB)
      gemm_f32_wmma<true><<<grid, 128, 0, stream>>>(A, lda, B, ldb, C, ldc, bias, M, Nn, K);
    else
      gemm_f32_wmma<false><<<grid, 128, 0, stream>>>(A, lda, B, ldb, C, ldc, bias, M, Nn, K);
  };

  auto run_gcn = [&](const float* g, int ld, int n, const float* h_in,
                     const float* bng, const float* bnb, const float* W,
                     const float* b, float* h_out) {
    bn_kernel<<<DIM, 256, 0, stream>>>(h_in, bng, bnb, bufA, n);
    gemm(bufA, DIM, W, DIM, bufB, DIM, b, n, DIM, DIM, true);
    rowsum_kernel<<<n, 256, 0, stream>>>(g, ld, n, vec1);
    vec_rsqrt_p1_kernel<<<(n + 255) / 256, 256, 0, stream>>>(vec1, vec2, n);
    rowscale_kernel<<<n, 256, 0, stream>>>(bufB, vec2, bufC, n);
    gemm(g, ld, bufC, DIM, bufD, DIM, nullptr, n, DIM, n, false);
    gcn_epi_kernel<<<n, 256, 0, stream>>>(bufD, bufC, vec2, h_out, n);
  };

  auto run_pool = [&](const float* g, int ld, int n, int kk, const float* h_in,
                      int level, float* g_next, float* h_next, int* idx_out) {
    bn_kernel<<<DIM, 256, 0, stream>>>(h_in, pool_bn_g + level * DIM,
                                       pool_bn_b + level * DIM, bufA, n);
    rowsum_kernel<<<n, 256, 0, stream>>>(g, ld, n, vec1);
    gemm(g, ld, bufA, DIM, bufB, DIM, nullptr, n, DIM, n, false);
    vec_recip_kernel<<<(n + 255) / 256, 256, 0, stream>>>(vec1, vec2, n);
    rowscale_kernel<<<n, 256, 0, stream>>>(bufB, vec2, bufC, n);
    pool_row_kernel<<<n, 256, 0, stream>>>(bufA, bufC, vec1, pool_W + level * DIM,
                                           pool_b, level, vec3, vec4, vec5);
    pool_scores_kernel<<<1, 256, 0, stream>>>(vec3, vec4, vec5, pool_sigma,
                                              level, n, vec6);
    topk_kernel<<<(n + 255) / 256, 256, 0, stream>>>(vec6, n, kk, idx_out);
    rowscale_kernel<<<n, 256, 0, stream>>>(bufA, vec6, bufD, n);
    {
      dim3 grid((n + 63) / 64, (kk + 63) / 64);
      bgemm_iu8_wmma<<<grid, 128, 0, stream>>>(g, ld, idx_out, big1, n, kk, n, n);
    }
    gemm(big1, n, bufD, DIM, h_next, DIM, nullptr, kk, DIM, n, false);
    {
      dim3 grid((kk + 255) / 256, kk);
      gather_cols_kernel<<<grid, 256, 0, stream>>>(big1, n, idx_out, big2, kk, kk);
    }
    rowsum_kernel<<<kk, 256, 0, stream>>>(big2, kk, kk, vec1);
    {
      dim3 grid((kk + 255) / 256, kk);
      coldiv_kernel<<<grid, 256, 0, stream>>>(big2, kk, g_next, kk, kk, vec1);
    }
  };

  auto run_cnn = [&](const float* h_scat, const float* h_pre, int level, int n,
                     float* out) {
    bn_kernel<<<DIM, 256, 0, stream>>>(h_scat, cnn_bn_g + level * DIM,
                                       cnn_bn_b + level * DIM, bufA, n);
    bn_kernel<<<DIM, 256, 0, stream>>>(h_pre, cnn_bn_g + level * DIM,
                                       cnn_bn_b + level * DIM, bufB, n);
    const float* W = cnn_W + (size_t)level * DIM * DIM;
    const float* b = cnn_b + level * DIM;
    gemm(bufA, DIM, W, DIM, bufC, DIM, b, n, DIM, DIM, true);
    gemm(bufB, DIM, W, DIM, bufD, DIM, b, n, DIM, DIM, true);
    gemm(bufC, DIM, bufC, DIM, big1, n, nullptr, n, n, DIM, true);
    softmax_row_kernel<<<n, 256, 0, stream>>>(big1, n, n);
    gemm(big1, n, bufD, DIM, bufF, DIM, nullptr, n, DIM, n, false);
    softmax_row_kernel<<<n, 256, 0, stream>>>(bufF, DIM, DIM);
    cnn_combine_kernel<<<n, 256, 0, stream>>>(bufF, bufD, bufC, out, n);
  };

  run_gcn(g0, N0, N0, h0, down_bn_g, down_bn_b, down_W, down_b, h_down0);
  run_pool(g0, N0, N0, KK0, h_down0, 0, g1, h_cur, idx0);
  run_gcn(g1, KK0, KK0, h_cur, down_bn_g + DIM, down_bn_b + DIM,
          down_W + DIM * DIM, down_b + DIM, h_down1);
  run_pool(g1, KK0, KK0, KK1, h_down1, 1, g2, h_cur, idx1);

  run_gcn(g2, KK1, KK1, h_cur, bot_bn_g, bot_bn_b, bot_W, bot_b, h_cur);

  fill_zero_kernel<<<(KK0 * DIM + 255) / 256, 256, 0, stream>>>(bufE, KK0 * DIM);
  scatter_rows_kernel<<<KK1, 256, 0, stream>>>(h_cur, idx1, bufE);
  run_cnn(bufE, h_down1, 0, KK0, bufG);
  run_gcn(g1, KK0, KK0, bufG, up_bn_g, up_bn_b, up_W, up_b, h_cur);

  fill_zero_kernel<<<(N0 * DIM + 255) / 256, 256, 0, stream>>>(bufE, N0 * DIM);
  scatter_rows_kernel<<<KK0, 256, 0, stream>>>(h_cur, idx0, bufE);
  run_cnn(bufE, h_down0, 1, N0, bufG);
  run_gcn(g0, N0, N0, bufG, up_bn_g + DIM, up_bn_b + DIM, up_W + DIM * DIM,
          up_b + DIM, h_cur);

  add_kernel<<<N0, 256, 0, stream>>>(h_cur, h0, (float*)d_out);
}
